// GATBlock_38062000177238
// MI455X (gfx1250) — matrix operand order repaired and checked by reading, not compiled, not run on hardware
//
#include <hip/hip_runtime.h>
#include <hip/hip_bf16.h>

typedef __attribute__((ext_vector_type(16))) _Float16 v16h;
typedef __attribute__((ext_vector_type(8)))  float    v8f;
typedef __attribute__((ext_vector_type(2)))  _Float16 v2h;
typedef __attribute__((ext_vector_type(4)))  unsigned int v4u;
typedef __attribute__((ext_vector_type(8)))  int      v8i_;
typedef __attribute__((ext_vector_type(4)))  int      v4i_;

#define N_NODES  50000
#define N_EDGES  800000
#define N_ETOT   850000   // E + N self loops
#define F_IN     128
#define HC       256      // H*C
#define NH       4
#define CH       64
#define C_OUT    64
#define NEG_SLOPE 0.2f
#define LN_EPS    1e-5f

// ---------------------------------------------------------------------------
// Helpers
// ---------------------------------------------------------------------------
__device__ __forceinline__ void atomicMaxFloat(float* addr, float val) {
    // int/uint punning trick: works for mixed signs, native atomics
    if (val >= 0.0f) atomicMax((int*)addr, __float_as_int(val));
    else             atomicMin((unsigned int*)addr, __float_as_uint(val));
}

__device__ __forceinline__ void atomicAddF32(float* addr, float val) {
    __hip_atomic_fetch_add(addr, val, __ATOMIC_RELAXED, __HIP_MEMORY_SCOPE_AGENT);
}

// ---------------------------------------------------------------------------
// Kernel 0: convert + transpose weights to f16 for WMMA B-fragments
//   wlrT  [512][128] : col-major view of [Wl | Wr]  (n-major, contiguous K)
//   wprojT[ 64][256] : Wproj^T
//   wresT [ 64][128] : Wres^T
// ---------------------------------------------------------------------------
__global__ void wconv_kernel(const float* __restrict__ Wl, const float* __restrict__ Wr,
                             const float* __restrict__ Wproj, const float* __restrict__ Wres,
                             _Float16* __restrict__ wlrT, _Float16* __restrict__ wprojT,
                             _Float16* __restrict__ wresT) {
    int tid = blockIdx.x * 256 + threadIdx.x;
    if (tid < 512 * 128) {
        int n = tid / 128, k = tid % 128;
        float v = (n < 256) ? Wl[k * 256 + n] : Wr[k * 256 + (n - 256)];
        wlrT[tid] = (_Float16)v;
    } else if (tid < 512 * 128 + 64 * 256) {
        int t = tid - 512 * 128;
        int n = t / 256, k = t % 256;
        wprojT[t] = (_Float16)Wproj[k * 64 + n];
    } else if (tid < 512 * 128 + 64 * 256 + 64 * 128) {
        int t = tid - 512 * 128 - 64 * 256;
        int n = t / 128, k = t % 128;
        wresT[t] = (_Float16)Wres[k * 64 + n];
    }
}

// ---------------------------------------------------------------------------
// Kernel 1: init emax=-inf, denom=0, acc=0  (must run every launch)
// ---------------------------------------------------------------------------
__global__ void init_kernel(float* __restrict__ emax, float* __restrict__ denom,
                            float* __restrict__ acc) {
    size_t tid = (size_t)blockIdx.x * 256 + threadIdx.x;
    if (tid < (size_t)N_NODES * HC) acc[tid] = 0.0f;
    if (tid < (size_t)N_NODES * NH) {
        emax[tid]  = -INFINITY;
        denom[tid] = 0.0f;
    }
}

// ---------------------------------------------------------------------------
// Kernel 2: node transforms  xl = x@Wl+bl,  xr = x@Wr+br  (f16 out) via WMMA.
// The 16x128 f32 A-tile is DMA'd into LDS by the Tensor Data Mover (wave 0
// builds the D# per ISA 8.3/8.4 and issues tensor_load_to_lds, then waits on
// TENSORcnt); all threads then convert f32->f16 in LDS. Also writes
// xh = f16(x) (blockIdx.y==0 only) for the residual GEMM.
// grid = (3125, 4), block = 256 (8 waves); wave w handles 16-col tile
// ct = blockIdx.y*8 + w over the 512 concatenated [Wl|Wr] columns.
// ---------------------------------------------------------------------------
__global__ void xform_kernel(const float* __restrict__ x,
                             const _Float16* __restrict__ wlrT,
                             const float* __restrict__ bl, const float* __restrict__ br,
                             _Float16* __restrict__ xlh, _Float16* __restrict__ xrh,
                             _Float16* __restrict__ xh) {
    __shared__ float    As32[16 * F_IN];  // 8 KB: TDM destination (raw f32 tile)
    __shared__ _Float16 As[16 * F_IN];    // 4 KB: converted f16 tile
    const int rowbase = blockIdx.x * 16;
    const int tid = threadIdx.x;

    if (tid < 32) {  // wave 0 issues the TDM descriptor
        // --- D# group 0: count=1 | lds_addr | 57-bit global_addr | type=2 ---
        unsigned long long gaddr =
            (unsigned long long)(const void*)x + (unsigned long long)rowbase * (F_IN * 4);
        // generic LDS pointer: low 32 bits are the workgroup-relative LDS offset
        unsigned lds_off = (unsigned)(unsigned long long)(void*)As32;
        v4u g0;
        g0[0] = 1u;                                           // count=1 (valid, user)
        g0[1] = lds_off;                                      // lds_addr
        g0[2] = (unsigned)(gaddr & 0xFFFFFFFFu);              // global_addr[31:0]
        g0[3] = (unsigned)((gaddr >> 32) & 0x01FFFFFFu)       // global_addr[56:32]
              | (2u << 30);                                   // type=2 ("image")
        // --- D# group 1: data_size=4B, 2D tensor 128 x 16, tile 128 x 16 ---
        v8i_ g1;
        g1[0] = (int)(2u << 16);          // workgroup_mask=0, data_size=2 (4 bytes)
        g1[1] = (int)(128u << 16);        // tensor_dim0[15:0] = 128 @ bits [63:48]
        g1[2] = (int)(16u << 16);         // tensor_dim1[15:0] = 16  @ bits [95:80]
        g1[3] = (int)(128u << 16);        // tile_dim0 = 128 @ bits [127:112]
        g1[4] = 16;                       // tile_dim1 = 16 @ bits [143:128]
        g1[5] = 128;                      // tensor_dim0_stride[31:0] = 128
        g1[6] = 0;
        g1[7] = 0;
        v4i_ gz = (v4i_){0, 0, 0, 0};
#if defined(__clang_major__) && (__clang_major__ >= 23)
        v8i_ gz8 = (v8i_){0, 0, 0, 0, 0, 0, 0, 0};
        __builtin_amdgcn_tensor_load_to_lds(g0, g1, gz, gz, gz8, 0);
#else
        __builtin_amdgcn_tensor_load_to_lds(g0, g1, gz, gz, 0);
#endif
        __builtin_amdgcn_s_wait_tensorcnt(0);
    }
    __syncthreads();

    // convert f32 tile -> f16 tile (and export xh once per row-tile)
    for (int i = tid; i < 16 * F_IN; i += 256) {
        float v = As32[i];
        As[i] = (_Float16)v;
        if (blockIdx.y == 0) {
            int r = i / F_IN, k = i % F_IN;
            xh[(size_t)(rowbase + r) * F_IN + k] = (_Float16)v;
        }
    }
    __syncthreads();

    const int wave = tid >> 5, lane = tid & 31;
    const int lg = lane >> 4, lm = lane & 15;
    const int colbase = (blockIdx.y * 8 + wave) * 16;  // 0..496

    v8f acc = {};
    #pragma unroll
    for (int kk = 0; kk < F_IN / 32; ++kk) {
        // A fragment: 16x32 f16, lane lm = row M, 16 contiguous K per lane
        v16h a;
        const _Float16* ap = &As[lm * F_IN + kk * 32 + lg * 16];
        #pragma unroll
        for (int i = 0; i < 16; ++i) a[i] = ap[i];
        // B fragment: 32x16 f16, lane lm = col N, 16 contiguous K per lane
        v16h b;
        const _Float16* bp = &wlrT[(size_t)(colbase + lm) * F_IN + kk * 32 + lg * 16];
        #pragma unroll
        for (int i = 0; i < 16; ++i) b[i] = bp[i];
        acc = __builtin_amdgcn_wmma_f32_16x16x32_f16(false, a, false, b,
                                                     (short)0, acc, false, false);
    }

    // uniform select of destination (colbase < 256 is wave-uniform):
    // avoids per-element exec-mask branching in the store stream
    const bool left = (colbase < HC);
    _Float16* __restrict__ dstp = left ? xlh : xrh;
    const float* __restrict__ biasp = left ? bl : br;
    const int coln = (colbase + lm) & (HC - 1);
    const float bias = biasp[coln];
    #pragma unroll
    for (int r = 0; r < 8; ++r) {
        int row = rowbase + r + 8 * lg;
        dstp[(size_t)row * HC + coln] = (_Float16)(acc[r] + bias);
    }
}

// ---------------------------------------------------------------------------
// Kernel 3: edge scores e[e,h] = sum_c lrelu(xl[src]+xr[dst]) * att[h,c]
// one thread per (edge, head); v2h/float2 vectorized; atomic max into emax
// ---------------------------------------------------------------------------
__global__ void score_kernel(const int* __restrict__ ei,
                             const _Float16* __restrict__ xlh,
                             const _Float16* __restrict__ xrh,
                             const float* __restrict__ att,
                             float* __restrict__ escore, float* __restrict__ emax) {
    size_t tid = (size_t)blockIdx.x * 256 + threadIdx.x;
    if (tid >= (size_t)N_ETOT * NH) return;
    int h = (int)(tid & 3);
    size_t e = tid >> 2;
    int src, dst;
    if (e < N_EDGES) { src = ei[e]; dst = ei[N_EDGES + e]; }
    else             { src = dst = (int)(e - N_EDGES); }

    const v2h*    pl = (const v2h*)   (xlh + (size_t)src * HC + h * CH);
    const v2h*    pr = (const v2h*)   (xrh + (size_t)dst * HC + h * CH);
    const float2* pa = (const float2*)(att + h * CH);
    __builtin_prefetch(pl, 0, 0);
    __builtin_prefetch(pr, 0, 0);

    float acc = 0.0f;
    #pragma unroll 8
    for (int c = 0; c < CH / 2; ++c) {
        v2h l = pl[c], r = pr[c];
        float2 a2 = pa[c];
        float m0 = (float)l[0] + (float)r[0];
        float m1 = (float)l[1] + (float)r[1];
        m0 = (m0 > 0.0f) ? m0 : NEG_SLOPE * m0;
        m1 = (m1 > 0.0f) ? m1 : NEG_SLOPE * m1;
        acc += m0 * a2.x + m1 * a2.y;
    }
    escore[e * NH + h] = acc;
    atomicMaxFloat(&emax[(size_t)dst * NH + h], acc);
}

// ---------------------------------------------------------------------------
// Kernel 4: ex = exp(e - emax[dst]); denom[dst,h] += ex;
//           acc[dst,h,c] += ex * xl[src,h,c]   (normalization folded out)
// ---------------------------------------------------------------------------
__global__ void msg_kernel(const int* __restrict__ ei,
                           const _Float16* __restrict__ xlh,
                           const float* __restrict__ escore,
                           const float* __restrict__ emax,
                           float* __restrict__ denom, float* __restrict__ acc) {
    size_t tid = (size_t)blockIdx.x * 256 + threadIdx.x;
    if (tid >= (size_t)N_ETOT * NH) return;
    int h = (int)(tid & 3);
    size_t e = tid >> 2;
    int src, dst;
    if (e < N_EDGES) { src = ei[e]; dst = ei[N_EDGES + e]; }
    else             { src = dst = (int)(e - N_EDGES); }

    float ex = __expf(escore[e * NH + h] - emax[(size_t)dst * NH + h]);
    atomicAddF32(&denom[(size_t)dst * NH + h], ex);

    const v2h* pl = (const v2h*)(xlh + (size_t)src * HC + h * CH);
    float* pd = acc + (size_t)dst * HC + h * CH;
    #pragma unroll 8
    for (int c = 0; c < CH / 2; ++c) {
        v2h l = pl[c];
        atomicAddF32(&pd[2 * c],     ex * (float)l[0]);
        atomicAddF32(&pd[2 * c + 1], ex * (float)l[1]);
    }
}

// ---------------------------------------------------------------------------
// Kernel 5: agg[n,hc] = acc[n,hc] / denom[n,h] + bias_out[hc]  -> f16
// ---------------------------------------------------------------------------
__global__ void agg_kernel(const float* __restrict__ acc, const float* __restrict__ denom,
                           const float* __restrict__ bias_out, _Float16* __restrict__ aggh) {
    size_t tid = (size_t)blockIdx.x * 256 + threadIdx.x;
    if (tid >= (size_t)N_NODES * HC) return;
    size_t n = tid / HC;
    int hc = (int)(tid % HC);
    float v = acc[tid] / denom[n * NH + (hc >> 6)] + bias_out[hc];
    aggh[tid] = (_Float16)v;
}

// ---------------------------------------------------------------------------
// Kernel 6: out = GELU(LN(agg @ Wproj)) + x @ Wres   via WMMA.
// One wave per 16-row tile computes all 4 col-tiles (64 outputs per row),
// LayerNorm is an intra-wave shfl_xor reduction over 16 lanes (wave32).
// block = 256 (8 waves), grid = ceil(3125/8) = 391.
// ---------------------------------------------------------------------------
__global__ void final_kernel(const _Float16* __restrict__ aggh,
                             const _Float16* __restrict__ xh,
                             const _Float16* __restrict__ wprojT,
                             const _Float16* __restrict__ wresT,
                             const float* __restrict__ ln_w, const float* __restrict__ ln_b,
                             float* __restrict__ out) {
    const int wave = threadIdx.x >> 5, lane = threadIdx.x & 31;
    const int rowbase = (blockIdx.x * 8 + wave) * 16;
    if (rowbase >= N_NODES) return;  // wave-uniform: EXEC stays all-ones
    const int lg = lane >> 4, lm = lane & 15;

    v8f d[4];
    #pragma unroll
    for (int t = 0; t < 4; ++t) d[t] = (v8f){};

    // ---- proj GEMM: [16x256] @ [256x64] ----
    #pragma unroll
    for (int kk = 0; kk < HC / 32; ++kk) {
        v16h a;
        const _Float16* ap = aggh + (size_t)(rowbase + lm) * HC + kk * 32 + lg * 16;
        #pragma unroll
        for (int i = 0; i < 16; ++i) a[i] = ap[i];
        #pragma unroll
        for (int t = 0; t < 4; ++t) {
            v16h b;
            const _Float16* bp = wprojT + (size_t)(t * 16 + lm) * HC + kk * 32 + lg * 16;
            #pragma unroll
            for (int i = 0; i < 16; ++i) b[i] = bp[i];
            d[t] = __builtin_amdgcn_wmma_f32_16x16x32_f16(false, a, false, b,
                                                          (short)0, d[t], false, false);
        }
    }

    // ---- LayerNorm over 64 cols (4 tiles x 16 lanes) + exact GELU ----
    float lnw[4], lnb[4];
    #pragma unroll
    for (int t = 0; t < 4; ++t) { lnw[t] = ln_w[t * 16 + lm]; lnb[t] = ln_b[t * 16 + lm]; }
    #pragma unroll
    for (int r = 0; r < 8; ++r) {
        float v0 = d[0][r], v1 = d[1][r], v2 = d[2][r], v3 = d[3][r];
        float s1 = v0 + v1 + v2 + v3;
        float s2 = v0 * v0 + v1 * v1 + v2 * v2 + v3 * v3;
        #pragma unroll
        for (int m = 1; m <= 8; m <<= 1) {   // xor over 16-lane group
            s1 += __shfl_xor(s1, m, 32);
            s2 += __shfl_xor(s2, m, 32);
        }
        float mu   = s1 * (1.0f / 64.0f);
        float var  = s2 * (1.0f / 64.0f) - mu * mu;
        float rstd = rsqrtf(var + LN_EPS);
        #pragma unroll
        for (int t = 0; t < 4; ++t) {
            float z = (d[t][r] - mu) * rstd * lnw[t] + lnb[t];
            d[t][r] = 0.5f * z * (1.0f + erff(z * 0.70710678118f));  // exact GELU
        }
    }

    // ---- residual GEMM: accumulate x @ Wres on top of GELU output ----
    #pragma unroll
    for (int kk = 0; kk < F_IN / 32; ++kk) {
        v16h a;
        const _Float16* ap = xh + (size_t)(rowbase + lm) * F_IN + kk * 32 + lg * 16;
        #pragma unroll
        for (int i = 0; i < 16; ++i) a[i] = ap[i];
        #pragma unroll
        for (int t = 0; t < 4; ++t) {
            v16h b;
            const _Float16* bp = wresT + (size_t)(t * 16 + lm) * F_IN + kk * 32 + lg * 16;
            #pragma unroll
            for (int i = 0; i < 16; ++i) b[i] = bp[i];
            d[t] = __builtin_amdgcn_wmma_f32_16x16x32_f16(false, a, false, b,
                                                          (short)0, d[t], false, false);
        }
    }

    // ---- store [16 x 64] f32 tile ----
    #pragma unroll
    for (int t = 0; t < 4; ++t) {
        int col = t * 16 + lm;
        #pragma unroll
        for (int r = 0; r < 8; ++r) {
            int row = rowbase + r + 8 * lg;
            out[(size_t)row * C_OUT + col] = d[t][r];
        }
    }
}

// ---------------------------------------------------------------------------
// Host-side orchestration
// ---------------------------------------------------------------------------
extern "C" void kernel_launch(void* const* d_in, const int* in_sizes, int n_in,
                              void* d_out, int out_size, void* d_ws, size_t ws_size,
                              hipStream_t stream) {
    const float* x        = (const float*)d_in[0];
    const int*   ei       = (const int*)  d_in[1];
    const float* Wl       = (const float*)d_in[2];
    const float* bl       = (const float*)d_in[3];
    const float* Wr       = (const float*)d_in[4];
    const float* br       = (const float*)d_in[5];
    const float* att      = (const float*)d_in[6];
    const float* bias_out = (const float*)d_in[7];
    const float* Wproj    = (const float*)d_in[8];
    const float* ln_w     = (const float*)d_in[9];
    const float* ln_b     = (const float*)d_in[10];
    const float* Wres     = (const float*)d_in[11];
    float* out = (float*)d_out;

    // workspace carve-up (256B aligned); total ~156 MB -> L2-resident pipeline
    char* wp = (char*)d_ws;
    auto carve = [&](size_t bytes) {
        char* p = wp;
        wp += (bytes + 255) & ~(size_t)255;
        return p;
    };
    _Float16* xlh    = (_Float16*)carve((size_t)N_NODES * HC * 2);
    _Float16* xrh    = (_Float16*)carve((size_t)N_NODES * HC * 2);
    _Float16* xh     = (_Float16*)carve((size_t)N_NODES * F_IN * 2);
    _Float16* aggh   = (_Float16*)carve((size_t)N_NODES * HC * 2);
    _Float16* wlrT   = (_Float16*)carve((size_t)512 * F_IN * 2);
    _Float16* wprojT = (_Float16*)carve((size_t)C_OUT * HC * 2);
    _Float16* wresT  = (_Float16*)carve((size_t)C_OUT * F_IN * 2);
    float*    escore = (float*)carve((size_t)N_ETOT * NH * 4);
    float*    emax   = (float*)carve((size_t)N_NODES * NH * 4);
    float*    denom  = (float*)carve((size_t)N_NODES * NH * 4);
    float*    acc    = (float*)carve((size_t)N_NODES * HC * 4);

    // 0: weight convert/transpose
    wconv_kernel<<<(512 * 128 + 64 * 256 + 64 * 128 + 255) / 256, 256, 0, stream>>>(
        Wl, Wr, Wproj, Wres, wlrT, wprojT, wresT);

    // 1: init scatter buffers (fresh every call)
    init_kernel<<<(N_NODES * HC + 255) / 256, 256, 0, stream>>>(emax, denom, acc);

    // 2: WMMA node transforms (A-tile staged by Tensor Data Mover)
    xform_kernel<<<dim3(N_NODES / 16, 4), 256, 0, stream>>>(x, wlrT, bl, br, xlh, xrh, xh);

    // 3: edge scores + segment max
    const int nEH = N_ETOT * NH;
    score_kernel<<<(nEH + 255) / 256, 256, 0, stream>>>(ei, xlh, xrh, att, escore, emax);

    // 4: exp + denom + message scatter
    msg_kernel<<<(nEH + 255) / 256, 256, 0, stream>>>(ei, xlh, escore, emax, denom, acc);

    // 5: normalize + bias -> f16
    agg_kernel<<<(N_NODES * HC + 255) / 256, 256, 0, stream>>>(acc, denom, bias_out, aggh);

    // 6: WMMA proj + LN + GELU + WMMA residual
    final_kernel<<<(N_NODES / 16 + 7) / 8, 256, 0, stream>>>(
        aggh, xh, wprojT, wresT, ln_w, ln_b, out);
}
